// DynamicSparseAttention_27702539059232
// MI455X (gfx1250) — matrix operand order repaired and checked by reading, not compiled, not run on hardware
//
#include <hip/hip_runtime.h>

// ---------- types ----------
typedef __attribute__((ext_vector_type(16))) __bf16  v16bf;
typedef __attribute__((ext_vector_type(8)))  float   v8f;
typedef __attribute__((ext_vector_type(4)))  unsigned int u32x4;

union ABu { v16bf v; u32x4 q[2]; };

__device__ __forceinline__ unsigned short f2bf_bits(float f) {
  union { float f; unsigned u; } a; a.f = f;
  unsigned u = a.u;
  unsigned r = u + 0x7FFFu + ((u >> 16) & 1u);   // round-to-nearest-even
  return (unsigned short)(r >> 16);
}
__device__ __forceinline__ float bf2f(unsigned short h) {
  union { float f; unsigned u; } a; a.u = ((unsigned)h) << 16;
  return a.f;
}

// ---------- problem constants ----------
#define BATCH 8
#define NSEQ  1024
#define CDIM  512
#define HEADS 8
#define DHEAD 64
#define TOPK  16
#define SCALE 0.125f   // 64^-0.5

// ---------- conversion kernels ----------
__global__ void cvt_f2bf(const float* __restrict__ in, unsigned short* __restrict__ out, int n) {
  int i = blockIdx.x * blockDim.x + threadIdx.x;
  if (i < n) out[i] = f2bf_bits(in[i]);
}

// WT[n*K + k] = bf16(W[k*Nc + n])   (transpose so WMMA B loads are contiguous)
__global__ void cvt_transpose(const float* __restrict__ W, unsigned short* __restrict__ WT,
                              int K, int Nc) {
  int i = blockIdx.x * blockDim.x + threadIdx.x;
  if (i < K * Nc) {
    int k = i / Nc, n = i - k * Nc;
    WT[(size_t)n * K + k] = f2bf_bits(W[i]);
  }
}

// ---------- generic bf16 WMMA GEMM: O = act(A[M,K] @ WT[Nc,K]^T + bias) ----------
// block = 128 threads (4 waves); each wave computes a 16(M) x 64(N) strip:
// 4 accumulators, A loaded once per k-step (4 WMMA per 10 b128 loads).
// grid = (Nc/64, M/64)
template <int RELU, int OUTBF16>
__global__ void gemm_wmma_bf16(const unsigned short* __restrict__ A,
                               const unsigned short* __restrict__ WT,
                               const float* __restrict__ bias,
                               unsigned short* __restrict__ outb,
                               float* __restrict__ outf,
                               int M, int K, int Nc) {
  int wave = threadIdx.x >> 5;
  int lane = threadIdx.x & 31;
  int half = lane >> 4;
  int r    = lane & 15;
  int n0 = blockIdx.x * 64;
  int m0 = blockIdx.y * 64 + wave * 16;

  const unsigned short* arow = A  + (size_t)(m0 + r) * K;
  const unsigned short* brow = WT + (size_t)(n0 + r) * K;
  const size_t bstride = (size_t)16 * K;     // next 16-col B tile

  v8f acc0 = {}, acc1 = {}, acc2 = {}, acc3 = {};
  for (int k0 = 0; k0 < K; k0 += 32) {
    ABu a, b0, b1, b2, b3;
    // A 16x32 bf16, this code's contiguous k order: lanes<16: row=r, K0..15; lanes>=16: row=r, K16..31
    a.q[0] = *(const u32x4*)(arow + k0 + half * 16);
    a.q[1] = *(const u32x4*)(arow + k0 + half * 16 + 8);
    // B 32x16 bf16, same contiguous k order as A: lanes<16: col=r, K0..15; lanes>=16: col=r, K16..31
    const unsigned short* bk = brow + k0 + half * 16;
    b0.q[0] = *(const u32x4*)(bk);                b0.q[1] = *(const u32x4*)(bk + 8);
    b1.q[0] = *(const u32x4*)(bk + bstride);      b1.q[1] = *(const u32x4*)(bk + bstride + 8);
    b2.q[0] = *(const u32x4*)(bk + 2 * bstride);  b2.q[1] = *(const u32x4*)(bk + 2 * bstride + 8);
    b3.q[0] = *(const u32x4*)(bk + 3 * bstride);  b3.q[1] = *(const u32x4*)(bk + 3 * bstride + 8);
    acc0 = __builtin_amdgcn_wmma_f32_16x16x32_bf16(false, a.v, false, b0.v, (short)0, acc0, false, false);
    acc1 = __builtin_amdgcn_wmma_f32_16x16x32_bf16(false, a.v, false, b1.v, (short)0, acc1, false, false);
    acc2 = __builtin_amdgcn_wmma_f32_16x16x32_bf16(false, a.v, false, b2.v, (short)0, acc2, false, false);
    acc3 = __builtin_amdgcn_wmma_f32_16x16x32_bf16(false, a.v, false, b3.v, (short)0, acc3, false, false);
  }

  v8f accs[4] = {acc0, acc1, acc2, acc3};
#pragma unroll
  for (int nt = 0; nt < 4; ++nt) {
    int col = n0 + nt * 16 + r;
    float bv = bias[col];
#pragma unroll
    for (int i = 0; i < 8; ++i) {
      int row = m0 + i + half * 8;        // C/D layout: VGPR i -> M=i (+8 for upper half)
      float v = accs[nt][i] + bv;
      if (RELU) v = v > 0.f ? v : 0.f;
      if (OUTBF16) outb[(size_t)row * Nc + col] = f2bf_bits(v);
      else         outf[(size_t)row * Nc + col] = v;
    }
  }
}

// ---------- per-(b,h) V column sums: vsum[b,h,d] = sum_n V[b,n,h,d] ----------
__global__ void vsum_kernel(const unsigned short* __restrict__ kvlin, float* __restrict__ vsum) {
  int id = blockIdx.x * blockDim.x + threadIdx.x;   // 4096 = B*H*DHEAD
  int d = id & 63;
  int h = (id >> 6) & 7;
  int b = id >> 9;
  const unsigned short* p = kvlin + ((size_t)b * NSEQ) * (2 * CDIM) + CDIM + h * DHEAD + d;
  float s = 0.f;
  for (int n = 0; n < NSEQ; ++n) s += bf2f(p[(size_t)n * (2 * CDIM)]);
  vsum[id] = s;
}

// ---------- fused scores + top-16 + softmax + sparse PV ----------
// block = 256 threads (8 waves); grid = B*H*(NSEQ/16)
__global__ void attn_topk_kernel(const unsigned short* __restrict__ qlin,
                                 const unsigned short* __restrict__ kvlin,
                                 const float* __restrict__ vsum,
                                 unsigned short* __restrict__ ctx) {
  __shared__ float S[16][NSEQ];        // 64 KB score tile (fits 320 KB WGP LDS)
  __shared__ float pvv[16][16];
  __shared__ int   pii[16][16];
  __shared__ float wval[16][TOPK];
  __shared__ int   widx[16][TOPK];
  __shared__ int   warg[16];
  __shared__ float rowM[16];
  __shared__ float rowSum[16];

  int bh = blockIdx.x >> 6;            // NSEQ/16 = 64 query tiles per (b,h)
  int qt = blockIdx.x & 63;
  int b = bh >> 3, h = bh & 7;
  int q0 = qt << 4;

  int tid  = threadIdx.x;
  int wave = tid >> 5;
  int lane = tid & 31;
  int half = lane >> 4;
  int r    = lane & 15;

  const size_t qbase = ((size_t)b * NSEQ + q0) * CDIM + h * DHEAD;

  // ---- Phase 1: S[16][1024] = relu(Q) @ K^T * scale  via WMMA ----
  const unsigned short* arow = qlin + qbase + (size_t)r * CDIM;
  for (int kt = wave; kt < NSEQ / 16; kt += 8) {
    int key = kt * 16 + r;
    const unsigned short* brow = kvlin + ((size_t)b * NSEQ + key) * (2 * CDIM) + h * DHEAD;
    v8f acc = {};
#pragma unroll
    for (int k0 = 0; k0 < DHEAD; k0 += 32) {
      ABu a, bb;
      a.q[0]  = *(const u32x4*)(arow + k0 + half * 16);
      a.q[1]  = *(const u32x4*)(arow + k0 + half * 16 + 8);
      bb.q[0] = *(const u32x4*)(brow + k0 + half * 16);
      bb.q[1] = *(const u32x4*)(brow + k0 + half * 16 + 8);
      acc = __builtin_amdgcn_wmma_f32_16x16x32_bf16(false, a.v, false, bb.v,
                                                    (short)0, acc, false, false);
    }
#pragma unroll
    for (int i = 0; i < 8; ++i)
      S[i + half * 8][kt * 16 + r] = acc[i] * SCALE;
  }
  __syncthreads();

  // ---- Phase 2: exact top-16 per row (values + indices) ----
  int row = tid >> 4;                  // 16 rows x 16 threads
  int t   = tid & 15;

  float bvl[TOPK];
  int   bil[TOPK];
#pragma unroll
  for (int j = 0; j < TOPK; ++j) { bvl[j] = -3.0e38f; bil[j] = 0; }
  for (int kk = 0; kk < 64; ++kk) {
    int key = (t << 6) + ((kk + 4 * t) & 63);   // rotate to spread LDS banks
    float v = S[row][key];
    if (v > bvl[TOPK - 1]) {
      int pos = TOPK - 1;
      while (pos > 0 && bvl[pos - 1] < v) {
        bvl[pos] = bvl[pos - 1]; bil[pos] = bil[pos - 1]; --pos;
      }
      bvl[pos] = v; bil[pos] = key;
    }
  }
  // 16 merge rounds: pop the global max of the 16 per-thread sorted lists
  int p = 0;
  for (int round = 0; round < TOPK; ++round) {
    pvv[row][t] = (p < TOPK) ? bvl[p] : -3.0e38f;
    pii[row][t] = (p < TOPK) ? bil[p] : 0;
    __syncthreads();
    if (t == 0) {
      float m = pvv[row][0]; int arg = 0;
      for (int j = 1; j < 16; ++j) if (pvv[row][j] > m) { m = pvv[row][j]; arg = j; }
      wval[row][round] = m;
      widx[row][round] = pii[row][arg];
      warg[row] = arg;
    }
    __syncthreads();
    if (t == warg[row]) ++p;
    __syncthreads();
  }

  // softmax over (top-16 values) U (1008 zeros)  — faithful to attn*mask semantics
  if (t == 0) {
    float M = wval[row][0];
    if (M < 0.f) M = 0.f;              // the masked zeros participate
    float s = 0.f;
    for (int j = 0; j < TOPK; ++j) s += __expf(wval[row][j] - M);
    s += (float)(NSEQ - TOPK) * __expf(-M);
    rowM[row] = M; rowSum[row] = s;
  }
  __syncthreads();

  // ---- Phase 3: out = p0 * sumV + sum_j (P_j - p0) * V[idx_j] ----
  float Mx = rowM[row], sm = rowSum[row];
  float p0 = __expf(-Mx) / sm;
  int d0 = t * 4;                      // each thread: 4 of the 64 head dims
  const float* vs = vsum + (size_t)bh * DHEAD + d0;
  float o[4];
#pragma unroll
  for (int i = 0; i < 4; ++i) o[i] = p0 * vs[i];
#pragma unroll
  for (int j = 0; j < TOPK; ++j) {
    int key = widx[row][j];
    float pj = __expf(wval[row][j] - Mx) / sm - p0;
    const unsigned short* vrow =
        kvlin + ((size_t)b * NSEQ + key) * (2 * CDIM) + CDIM + h * DHEAD + d0;
#pragma unroll
    for (int i = 0; i < 4; ++i) o[i] += pj * bf2f(vrow[i]);
  }
  unsigned short* crow = ctx + ((size_t)b * NSEQ + q0 + row) * CDIM + h * DHEAD + d0;
#pragma unroll
  for (int i = 0; i < 4; ++i) crow[i] = f2bf_bits(o[i]);
}

// ---------- launch ----------
extern "C" void kernel_launch(void* const* d_in, const int* in_sizes, int n_in,
                              void* d_out, int out_size, void* d_ws, size_t ws_size,
                              hipStream_t stream) {
  const float* x   = (const float*)d_in[0];
  const float* Wq  = (const float*)d_in[1];
  const float* bq  = (const float*)d_in[2];
  const float* Wkv = (const float*)d_in[3];
  const float* bkv = (const float*)d_in[4];
  const float* Wp  = (const float*)d_in[5];
  const float* bp  = (const float*)d_in[6];
  float* out = (float*)d_out;

  const int M = BATCH * NSEQ;          // 8192
  char* ws = (char*)d_ws;
  size_t off = 0;
  auto alloc = [&](size_t bytes) { char* p = ws + off; off += (bytes + 255) & ~(size_t)255; return p; };

  unsigned short* xb    = (unsigned short*)alloc((size_t)M * CDIM * 2);          // 8 MB
  unsigned short* wtq   = (unsigned short*)alloc((size_t)CDIM * CDIM * 2);       // 0.5 MB
  unsigned short* wtkv  = (unsigned short*)alloc((size_t)CDIM * 2 * CDIM * 2);   // 1 MB
  unsigned short* wtp   = (unsigned short*)alloc((size_t)CDIM * CDIM * 2);       // 0.5 MB
  unsigned short* qlin  = (unsigned short*)alloc((size_t)M * CDIM * 2);          // 8 MB
  unsigned short* kvlin = (unsigned short*)alloc((size_t)M * 2 * CDIM * 2);      // 16 MB
  unsigned short* ctx   = (unsigned short*)alloc((size_t)M * CDIM * 2);          // 8 MB
  float*          vsum  = (float*)alloc((size_t)BATCH * HEADS * DHEAD * 4);      // 16 KB

  // 1. convert inputs to bf16 (weights transposed to [N][K])
  {
    int n = M * CDIM;
    cvt_f2bf<<<(n + 255) / 256, 256, 0, stream>>>(x, xb, n);
    cvt_transpose<<<(CDIM * CDIM + 255) / 256, 256, 0, stream>>>(Wq, wtq, CDIM, CDIM);
    cvt_transpose<<<(CDIM * 2 * CDIM + 255) / 256, 256, 0, stream>>>(Wkv, wtkv, CDIM, 2 * CDIM);
    cvt_transpose<<<(CDIM * CDIM + 255) / 256, 256, 0, stream>>>(Wp, wtp, CDIM, CDIM);
  }

  // 2. q = relu(x@Wq + bq)  (bf16 out) ; kv = x@Wkv + bkv (bf16 out)
  gemm_wmma_bf16<1, 1><<<dim3(CDIM / 64, M / 64), 128, 0, stream>>>(
      xb, wtq, bq, qlin, nullptr, M, CDIM, CDIM);
  gemm_wmma_bf16<0, 1><<<dim3(2 * CDIM / 64, M / 64), 128, 0, stream>>>(
      xb, wtkv, bkv, kvlin, nullptr, M, CDIM, 2 * CDIM);

  // 3. V column sums per (b,h)
  vsum_kernel<<<(BATCH * HEADS * DHEAD) / 256, 256, 0, stream>>>(kvlin, vsum);

  // 4. fused scores -> top-16 -> softmax -> sparse PV
  attn_topk_kernel<<<BATCH * HEADS * (NSEQ / 16), 256, 0, stream>>>(qlin, kvlin, vsum, ctx);

  // 5. out = ctx @ Wp + bp  (fp32 out)
  gemm_wmma_bf16<0, 0><<<dim3(CDIM / 64, M / 64), 128, 0, stream>>>(
      ctx, wtp, bp, nullptr, out, M, CDIM, CDIM);
}
